// LSTM_22608707846149
// MI455X (gfx1250) — matrix operand order repaired and checked by reading, not compiled, not run on hardware
//
#include <hip/hip_runtime.h>

// ---------------------------------------------------------------------------
// Types for CDNA5 WMMA (wave32): D(16x16,f32) = A(16x32,bf16) x B(32x16,bf16) + C
// ---------------------------------------------------------------------------
typedef __attribute__((ext_vector_type(16))) __bf16 v16bf;
typedef __attribute__((ext_vector_type(8)))  float  v8f;

union FragU { v16bf v; uint4 u[2]; };

__device__ __forceinline__ unsigned short f2bf(float f) {
    unsigned int u = __float_as_uint(f);
    unsigned int r = u + 0x7FFFu + ((u >> 16) & 1u);   // round-to-nearest-even
    return (unsigned short)(r >> 16);
}
__device__ __forceinline__ float bf2f(unsigned short b) {
    return __uint_as_float(((unsigned int)b) << 16);
}
__device__ __forceinline__ float sigm(float x) { return 1.0f / (1.0f + __expf(-x)); }
__device__ __forceinline__ float tanh_f(float x) {
    float e = __expf(-2.0f * fabsf(x));
    float r = (1.0f - e) / (1.0f + e);
    return copysignf(r, x);
}

// ---------------------------------------------------------------------------
// Fragment-swizzled layouts (so every hot-loop lane load is base + lane*32B).
//
// A-matrix (16x32 bf16 tile) ISA layout:
//   lane<16 : M=lane,    K = {0..7, 16..23}
//   lane>=16: M=lane-16, K = {8..15, 24..31}
// Swizzled A buffer: [mtile][ktile][lane][16 bf16], KT = #ktiles of the buffer.
// ---------------------------------------------------------------------------
__device__ __forceinline__ size_t swzA(int KT, int row, int k) {
    int mt = row >> 4, lr = row & 15;
    int kt = k >> 5,  ko = k & 31;
    int hl  = (ko >> 3) & 1;        // which half-wave holds this K
    int l   = lr + (hl << 4);
    int run = ko >> 4;              // first or second 8-element run
    int e   = ko & 7;
    return ((((size_t)mt * KT + kt) * 32 + l) << 4) + (size_t)(run * 8 + e);
}

// B-matrix (32x16 bf16 tile): lane<16: N=lane, K={0..7,16..23}; lane>=16: N=lane-16, K={8..15,24..31}.
// Swizzled WT buffer: [ktile][ntile(256)][lane][16 bf16].
__device__ __forceinline__ size_t swzB(int k, int n) {
    return ((((size_t)(k >> 5) * 256 + (n >> 4)) * 32 + ((n & 15) + (((k >> 3) & 1) << 4))) << 4) + (size_t)((((k & 31) >> 4) << 3) + (k & 7));
}

__device__ __forceinline__ v16bf load_a_frag(const unsigned short* __restrict__ base,
                                             int KT, int m0, int k0, int lane) {
    const unsigned short* p =
        base + ((((size_t)(m0 >> 4) * KT + (k0 >> 5)) * 32 + lane) << 4);
    FragU f;
    f.u[0] = *(const uint4*)(p);
    f.u[1] = *(const uint4*)(p + 8);
    return f.v;
}

__device__ __forceinline__ v16bf load_b_frag(const unsigned short* __restrict__ wt,
                                             int k0, int ncol, int lane) {
    const unsigned short* p =
        wt + ((((size_t)(k0 >> 5) * 256 + (ncol >> 4)) * 32 + lane) << 4);
    __builtin_prefetch(p + 131072, 0, 1);   // next K-tile (256 KB ahead) of weight stream
    FragU f;
    f.u[0] = *(const uint4*)(p);
    f.u[1] = *(const uint4*)(p + 8);
    return f.v;
}

__device__ __forceinline__ v8f wmma_bf16(v16bf a, v16bf b, v8f c) {
    return __builtin_amdgcn_wmma_f32_16x16x32_bf16(false, a, false, b, (short)0, c, false, false);
}

// ---------------------------------------------------------------------------
// Prep kernels
// ---------------------------------------------------------------------------
// src: W [4096, K] row-major (f32). dst: swizzled bf16 WT with global k' = k_off+k.
__global__ void transpose_cast_kernel(const float* __restrict__ src,
                                      unsigned short* __restrict__ dst,
                                      int K, int k_off) {
    long i = (long)blockIdx.x * blockDim.x + threadIdx.x;
    long total = 4096L * K;
    if (i >= total) return;
    int n = (int)(i / K);
    int k = (int)(i % K);
    dst[swzB(k_off + k, n)] = f2bf(src[(size_t)n * K + k]);
}

// x [B=256, T=128, F=256] f32 -> per-timestep swizzled bf16 A tiles (KT=8).
__global__ void cast_x_kernel(const float* __restrict__ src, unsigned short* __restrict__ dst) {
    long i = (long)blockIdx.x * blockDim.x + threadIdx.x;
    if (i >= 256L * 128 * 256) return;
    int m = (int)(i >> 15);            // / (128*256)
    int t = (int)((i >> 8) & 127);
    int k = (int)(i & 255);
    dst[(size_t)t * 65536 + swzA(8, m, k)] = f2bf(src[i]);
}

__global__ void bias_sum_kernel(const float* __restrict__ b1, const float* __restrict__ b2,
                                float* __restrict__ out, int n) {
    int i = blockIdx.x * blockDim.x + threadIdx.x;
    if (i < n) out[i] = b1[i] + b2[i];
}

__global__ void zero_state_kernel(unsigned short* __restrict__ h, float* __restrict__ c, int n) {
    int i = blockIdx.x * blockDim.x + threadIdx.x;
    if (i < n) { h[i] = 0; c[i] = 0.0f; }
}

// ---------------------------------------------------------------------------
// Fused LSTM step: gates = [in_t | h_{t-1}] @ WT + bias ; cell update in-register.
// Wave -> 16 h-columns (n0..n0+15), all 4 gate tiles, MW=2 M-tiles (32 batch rows).
// Grid: x = 1024/(4*16) = 16 blocks (4 waves/block), y = 256/32 = 8 blocks.
// ---------------------------------------------------------------------------
#define MW 2

__global__ __launch_bounds__(128) void lstm_step_kernel(
    const unsigned short* __restrict__ Ain,    // swizzled bf16 A for this timestep
    int KTin,                                  // #ktiles of Ain (8 for x, 32 for h1seq)
    int K_in,                                  // input width (256 or 1024)
    const unsigned short* __restrict__ WT,     // swizzled bf16 weights
    int K_total,
    const float* __restrict__ bias,            // [4096] = bih + bhh
    unsigned short* __restrict__ h_state,      // swizzled bf16, KT=32 (256x1024)
    float* __restrict__ c_state,               // f32 [256,1024] linear
    unsigned short* __restrict__ hseq_out)     // swizzled bf16 slice for t, or nullptr
{
    const int lane = threadIdx.x & 31;
    const int wave = threadIdx.x >> 5;
    const int n0   = (blockIdx.x * 4 + wave) * 16;   // h column tile
    const int m0   = blockIdx.y * (16 * MW);         // batch row tile

    v8f acc[MW][4];
    const v8f vzero = {};
#pragma unroll
    for (int mi = 0; mi < MW; ++mi)
#pragma unroll
        for (int g = 0; g < 4; ++g) acc[mi][g] = vzero;

    // ---- phase 1: input projection ----
    for (int kk = 0; kk < K_in; kk += 32) {
        v16bf b0 = load_b_frag(WT, kk, n0,        lane);
        v16bf b1 = load_b_frag(WT, kk, n0 + 1024, lane);
        v16bf b2 = load_b_frag(WT, kk, n0 + 2048, lane);
        v16bf b3 = load_b_frag(WT, kk, n0 + 3072, lane);
#pragma unroll
        for (int mi = 0; mi < MW; ++mi) {
            v16bf a = load_a_frag(Ain, KTin, m0 + 16 * mi, kk, lane);
            acc[mi][0] = wmma_bf16(a, b0, acc[mi][0]);
            acc[mi][1] = wmma_bf16(a, b1, acc[mi][1]);
            acc[mi][2] = wmma_bf16(a, b2, acc[mi][2]);
            acc[mi][3] = wmma_bf16(a, b3, acc[mi][3]);
        }
    }

    // ---- phase 2: recurrent projection (A = h_{t-1}, KT=32) ----
    for (int kk = K_in; kk < K_total; kk += 32) {
        v16bf b0 = load_b_frag(WT, kk, n0,        lane);
        v16bf b1 = load_b_frag(WT, kk, n0 + 1024, lane);
        v16bf b2 = load_b_frag(WT, kk, n0 + 2048, lane);
        v16bf b3 = load_b_frag(WT, kk, n0 + 3072, lane);
#pragma unroll
        for (int mi = 0; mi < MW; ++mi) {
            v16bf a = load_a_frag(h_state, 32, m0 + 16 * mi, kk - K_in, lane);
            acc[mi][0] = wmma_bf16(a, b0, acc[mi][0]);
            acc[mi][1] = wmma_bf16(a, b1, acc[mi][1]);
            acc[mi][2] = wmma_bf16(a, b2, acc[mi][2]);
            acc[mi][3] = wmma_bf16(a, b3, acc[mi][3]);
        }
    }

    // ---- fused LSTM cell epilogue ----
    // C/D layout: lane<16 -> N=lane, M=r ; lane>=16 -> N=lane-16, M=r+8 (per VGPR r)
    const int col = n0 + (lane & 15);
    const float bi = bias[col];
    const float bf = bias[col + 1024];
    const float bg = bias[col + 2048];
    const float bo = bias[col + 3072];
    const int mrow_hi = (lane >> 4) << 3;

#pragma unroll
    for (int mi = 0; mi < MW; ++mi) {
#pragma unroll
        for (int r = 0; r < 8; ++r) {
            const int m = m0 + 16 * mi + r + mrow_hi;
            const size_t cidx = (size_t)m * 1024 + col;
            float gi = sigm(acc[mi][0][r] + bi);
            float gf = sigm(acc[mi][1][r] + bf);
            float gg = tanh_f(acc[mi][2][r] + bg);
            float go = sigm(acc[mi][3][r] + bo);
            float c  = gf * c_state[cidx] + gi * gg;
            c_state[cidx] = c;
            float h = go * tanh_f(c);
            unsigned short hb = f2bf(h);
            const size_t hidx = swzA(32, m, col);
            h_state[hidx] = hb;
            if (hseq_out) hseq_out[hidx] = hb;
        }
    }
}

// ---------------------------------------------------------------------------
// Final FC: out[256,2] = h_last @ Wfc^T + bfc   (h_last is swizzled, KT=32)
// ---------------------------------------------------------------------------
__global__ void fc_kernel(const unsigned short* __restrict__ h,
                          const float* __restrict__ Wfc,          // [2,1024]
                          const float* __restrict__ bfc,          // [2]
                          float* __restrict__ out) {              // [256,2]
    int idx = blockIdx.x * blockDim.x + threadIdx.x;
    if (idx >= 256 * 2) return;
    int m = idx >> 1, o = idx & 1;
    float s = bfc[o];
    for (int k = 0; k < 1024; ++k)
        s += bf2f(h[swzA(32, m, k)]) * Wfc[o * 1024 + k];
    out[idx] = s;
}

// ---------------------------------------------------------------------------
// Host launcher
// ---------------------------------------------------------------------------
extern "C" void kernel_launch(void* const* d_in, const int* in_sizes, int n_in,
                              void* d_out, int out_size, void* d_ws, size_t ws_size,
                              hipStream_t stream) {
    const float* x    = (const float*)d_in[0];
    const float* Wih0 = (const float*)d_in[1];
    const float* Whh0 = (const float*)d_in[2];
    const float* bih0 = (const float*)d_in[3];
    const float* bhh0 = (const float*)d_in[4];
    const float* Wih1 = (const float*)d_in[5];
    const float* Whh1 = (const float*)d_in[6];
    const float* bih1 = (const float*)d_in[7];
    const float* bhh1 = (const float*)d_in[8];
    const float* Wfc  = (const float*)d_in[9];
    const float* bfc  = (const float*)d_in[10];
    float* out = (float*)d_out;

    // Workspace carve-up (~108 MB)
    char* ws = (char*)d_ws;
    unsigned short* WT0   = (unsigned short*)ws;  ws += 1280L * 4096 * 2;   // swz bf16
    unsigned short* WT1   = (unsigned short*)ws;  ws += 2048L * 4096 * 2;   // swz bf16
    unsigned short* xbf   = (unsigned short*)ws;  ws += 256L * 128 * 256 * 2;  // swz per t
    unsigned short* h1seq = (unsigned short*)ws;  ws += 256L * 128 * 1024 * 2; // swz per t
    float* bias0          = (float*)ws;           ws += 4096 * 4;
    float* bias1          = (float*)ws;           ws += 4096 * 4;
    unsigned short* hst   = (unsigned short*)ws;  ws += 256L * 1024 * 2;    // swz, KT=32
    float* cst            = (float*)ws;           ws += 256L * 1024 * 4;    // linear

    // --- prep ---
    {
        long t0 = 4096L * 256;
        transpose_cast_kernel<<<(unsigned)((t0 + 255) / 256), 256, 0, stream>>>(Wih0, WT0, 256, 0);
        long t1 = 4096L * 1024;
        transpose_cast_kernel<<<(unsigned)((t1 + 255) / 256), 256, 0, stream>>>(Whh0, WT0, 1024, 256);
        transpose_cast_kernel<<<(unsigned)((t1 + 255) / 256), 256, 0, stream>>>(Wih1, WT1, 1024, 0);
        transpose_cast_kernel<<<(unsigned)((t1 + 255) / 256), 256, 0, stream>>>(Whh1, WT1, 1024, 1024);
        long nx = 256L * 128 * 256;
        cast_x_kernel<<<(unsigned)((nx + 255) / 256), 256, 0, stream>>>(x, xbf);
        bias_sum_kernel<<<16, 256, 0, stream>>>(bih0, bhh0, bias0, 4096);
        bias_sum_kernel<<<16, 256, 0, stream>>>(bih1, bhh1, bias1, 4096);
        zero_state_kernel<<<1024, 256, 0, stream>>>(hst, cst, 256 * 1024);
    }

    dim3 sgrid(16, 8), sblock(128);

    // --- layer 0: K_in = 256 (x, KT=8), K_total = 1280 ---
    for (int t = 0; t < 128; ++t) {
        lstm_step_kernel<<<sgrid, sblock, 0, stream>>>(
            xbf + (size_t)t * 65536, 8, 256,
            WT0, 1280, bias0, hst, cst,
            h1seq + (size_t)t * 262144);
    }

    zero_state_kernel<<<1024, 256, 0, stream>>>(hst, cst, 256 * 1024);

    // --- layer 1: K_in = 1024 (h1seq, KT=32), K_total = 2048 ---
    for (int t = 0; t < 128; ++t) {
        lstm_step_kernel<<<sgrid, sblock, 0, stream>>>(
            h1seq + (size_t)t * 262144, 32, 1024,
            WT1, 2048, bias1, hst, cst,
            nullptr);
    }

    // --- FC head ---
    fc_kernel<<<2, 256, 0, stream>>>(hst, Wfc, bfc, out);
}
